// QuantizedLinear_31095563223127
// MI455X (gfx1250) — matrix operand order repaired and checked by reading, not compiled, not run on hardware
//
#include <hip/hip_runtime.h>
#include <hip/hip_bf16.h>

typedef __attribute__((ext_vector_type(16))) _Float16 v16h;
typedef __attribute__((ext_vector_type(8)))  _Float16 v8h;
typedef __attribute__((ext_vector_type(4)))  _Float16 v4h;
typedef __attribute__((ext_vector_type(8)))  float    v8f;
typedef __attribute__((ext_vector_type(4)))  float    v4f;
typedef __attribute__((ext_vector_type(4)))  int      v4i;
typedef __attribute__((ext_vector_type(4)))  unsigned int u32x4;
typedef __attribute__((ext_vector_type(8)))  int      i32x8;
typedef __attribute__((ext_vector_type(4)))  int      i32x4;

static constexpr int OUT_F = 4096;
static constexpr int IN_F  = 4096;
static constexpr int MDIM  = 8192;          // 4 * 2048
static constexpr int KDIM  = IN_F;          // 4096
static constexpr int NDIM  = OUT_F;         // 4096
static constexpr int QBLK  = 1024;
static constexpr int NQBLK = OUT_F * IN_F / QBLK;   // 16384

// Use the Tensor Data Mover when the toolchain exposes it.
#if defined(__has_builtin)
#if __has_builtin(__builtin_amdgcn_tensor_load_to_lds)
#define USE_TDM 1
#endif
#endif
// The therock-10.0 headers (6-arg builtin toolchain) ship this header.
#if defined(__has_include)
#if __has_include(<hip/amd_detail/amd_gfx1250_TDM.h>)
#define TDM_6ARG 1
#endif
#endif

__constant__ float c_nf4[16] = {
    -1.0f, -0.6962f, -0.5251f, -0.3949f, -0.2844f, -0.1848f, -0.0911f, 0.0f,
     0.0796f, 0.1609f, 0.2461f, 0.3379f, 0.4407f, 0.5626f, 0.723f, 1.0f};

// ---------------------------------------------------------------------------
// Kernel 1: NF4 dequant -> f16 weight matrix W16[NDIM][KDIM] in workspace.
// ---------------------------------------------------------------------------
__global__ __launch_bounds__(256) void nf4_dequant_kernel(
    const int* __restrict__ qidx, const float* __restrict__ scales,
    _Float16* __restrict__ w16) {
  __shared__ float lut[16];
  if (threadIdx.x < 16) lut[threadIdx.x] = c_nf4[threadIdx.x];
  __syncthreads();

  const int blk = blockIdx.x;
  const float sc = scales[blk];                     // uniform -> s_load
  const size_t base = (size_t)blk * QBLK + (size_t)threadIdx.x * 4;

  v4i idx = *(const v4i*)(qidx + base);
  v4h o;
  o.x = (_Float16)(lut[idx.x & 15] * sc);
  o.y = (_Float16)(lut[idx.y & 15] * sc);
  o.z = (_Float16)(lut[idx.z & 15] * sc);
  o.w = (_Float16)(lut[idx.w & 15] * sc);
  *(v4h*)(w16 + base) = o;
}

// ---------------------------------------------------------------------------
// Kernel 2: x f32 -> f16, 8 elements per thread.
// ---------------------------------------------------------------------------
__global__ __launch_bounds__(256) void cvt_f32_to_f16_kernel(
    const float* __restrict__ x, _Float16* __restrict__ y) {
  const size_t i = ((size_t)blockIdx.x * blockDim.x + threadIdx.x) * 8;
  v4f a = *(const v4f*)(x + i);
  v4f b = *(const v4f*)(x + i + 4);
  v8h h;
  h[0] = (_Float16)a.x; h[1] = (_Float16)a.y;
  h[2] = (_Float16)a.z; h[3] = (_Float16)a.w;
  h[4] = (_Float16)b.x; h[5] = (_Float16)b.y;
  h[6] = (_Float16)b.z; h[7] = (_Float16)b.w;
  *(v8h*)(y + i) = h;
}

// ---------------------------------------------------------------------------
// TDM descriptor issue: load a 128-row x 32-half tile (memory row stride =
// KDIM halfs) into LDS with +4-DWORD padding per 16-DWORD row so the LDS
// row stride is 20 DWORDs = 40 halfs (matches LDT below).
//   D# group0: count=1 | lds_addr | global_addr | type=2
//   D# group1: data_size=1(2B), pad_enable, pad_interval=3 (16 DW),
//              pad_amount=3 (4 DW), tensor_dim0=K, tensor_dim1=rows,
//              tile_dim0=32, tile_dim1=128, dim0_stride=K
// ---------------------------------------------------------------------------
#if defined(USE_TDM)
__device__ __forceinline__ void tdm_load_tile(unsigned int lds_byte_off,
                                              unsigned long long gaddr,
                                              unsigned int rows) {
  u32x4 g0;
  g0.x = 1u;                                        // count=1, user descriptor
  g0.y = lds_byte_off;                              // bits 63:32
  g0.z = (unsigned int)gaddr;                       // global_addr[31:0]
  g0.w = (unsigned int)(gaddr >> 32) | (2u << 30);  // addr[56:32] | type=2
  i32x8 g1;
  g1[0] = (1 << 16)      // data_size = 2 bytes
        | (1 << 20)      // pad_enable
        | (3 << 22)      // pad_interval: 16 DWORDs
        | (3 << 25);     // pad_amount: 4 DWORDs
  g1[1] = (int)(((unsigned)KDIM & 0xFFFFu) << 16);            // dim0 lo16
  g1[2] = (int)(((unsigned)KDIM >> 16) | ((rows & 0xFFFFu) << 16)); // dim0 hi | dim1 lo
  g1[3] = (int)((rows >> 16) | (32u << 16));                  // dim1 hi | tile_dim0=32
  g1[4] = 128;                                                // tile_dim1=128
  g1[5] = KDIM;                                               // dim0_stride lo32
  g1[6] = 0;
  g1[7] = 0;
  i32x4 z4 = {0, 0, 0, 0};
#if defined(TDM_6ARG)
  i32x8 z8 = {0, 0, 0, 0, 0, 0, 0, 0};
  __builtin_amdgcn_tensor_load_to_lds(g0, g1, z4, z4, z8, 0);
#else
  __builtin_amdgcn_tensor_load_to_lds(g0, g1, z4, z4, 0);
#endif
}
#endif

// ---------------------------------------------------------------------------
// Kernel 3: WMMA GEMM.  C[M][N] = A[M][K] * B[N][K]^T, all K-contiguous (TN).
// Block tile 128x128, K-step 32, 8 waves (2x4), 8 WMMA / wave / step.
// 3-stage circular LDS pipeline: TDM for tile t+2 issued at iter t, so each
// DMA gets ~2 compute iterations of latency slack; per-wave in-order TDM
// completion means "TENSORcnt <= 1" proves tile t+1 is resident.
// ---------------------------------------------------------------------------
__global__ __launch_bounds__(256) void nf4_wmma_gemm_kernel(
    const _Float16* __restrict__ A,   // [MDIM, KDIM] f16
    const _Float16* __restrict__ B,   // [NDIM, KDIM] f16
    float* __restrict__ C) {          // [MDIM, NDIM] f32
  constexpr int BM = 128, BN = 128, BK = 32;
  constexpr int LDT = 40;             // padded halfs per LDS row
  constexpr int STG = 3;              // pipeline stages
  __shared__ __align__(16) _Float16 sA[STG][BM * LDT];   // 30,720 B
  __shared__ __align__(16) _Float16 sB[STG][BN * LDT];   // 30,720 B

  const int tid  = threadIdx.x;
  const int lane = tid & 31;
  const int lh   = lane & 15;         // lane within half-wave
  const int hi   = lane >> 4;         // half-wave select (0/1)
  const int wave = tid >> 5;          // 0..7
  const int wm   = wave & 1;          // 2 waves along M (64 rows each)
  const int wn   = wave >> 1;         // 4 waves along N (32 cols each)
  const int m0   = blockIdx.y * BM;
  const int n0   = blockIdx.x * BN;

  v8f acc[4][2] = {};
  constexpr int NT = KDIM / BK;       // 128 K-steps

#if defined(USE_TDM)
  // Flat LDS pointers: low 32 bits of the generic address = LDS byte offset.
  const unsigned int ldsA[STG] = {(unsigned int)(uintptr_t)&sA[0][0],
                                  (unsigned int)(uintptr_t)&sA[1][0],
                                  (unsigned int)(uintptr_t)&sA[2][0]};
  const unsigned int ldsB[STG] = {(unsigned int)(uintptr_t)&sB[0][0],
                                  (unsigned int)(uintptr_t)&sB[1][0],
                                  (unsigned int)(uintptr_t)&sB[2][0]};
  const unsigned long long gA =
      (unsigned long long)(uintptr_t)A + (unsigned long long)m0 * KDIM * 2ull;
  const unsigned long long gB =
      (unsigned long long)(uintptr_t)B + (unsigned long long)n0 * KDIM * 2ull;
  constexpr unsigned long long KB = (unsigned long long)BK * 2ull; // bytes/step

  // Prologue: issue tiles 0 and 1; ensure tile 0 resident (<=1 outstanding).
  if (wave == 0) {
    tdm_load_tile(ldsA[0], gA, (unsigned int)MDIM);
    tdm_load_tile(ldsA[1], gA + KB, (unsigned int)MDIM);
  } else if (wave == 1) {
    tdm_load_tile(ldsB[0], gB, (unsigned int)NDIM);
    tdm_load_tile(ldsB[1], gB + KB, (unsigned int)NDIM);
  }
  __builtin_amdgcn_s_wait_tensorcnt((short)1);
  __syncthreads();
#else
  // Manual staging fallback: 512 chunks of 16B per matrix; 2 per thread.
  const int row0 = tid >> 2;
  const int row1 = (tid + 256) >> 2;
  const int kp   = (tid & 3) * 8;
  const _Float16* gA0 = A + (size_t)(m0 + row0) * KDIM + kp;
  const _Float16* gA1 = A + (size_t)(m0 + row1) * KDIM + kp;
  const _Float16* gB0 = B + (size_t)(n0 + row0) * KDIM + kp;
  const _Float16* gB1 = B + (size_t)(n0 + row1) * KDIM + kp;

  v8h ra0 = *(const v8h*)gA0;
  v8h ra1 = *(const v8h*)gA1;
  v8h rb0 = *(const v8h*)gB0;
  v8h rb1 = *(const v8h*)gB1;
  *(v8h*)&sA[0][row0 * LDT + kp] = ra0;
  *(v8h*)&sA[0][row1 * LDT + kp] = ra1;
  *(v8h*)&sB[0][row0 * LDT + kp] = rb0;
  *(v8h*)&sB[0][row1 * LDT + kp] = rb1;
  __syncthreads();
#endif

  int cur = 0;                        // buffer holding tile kt
  int ins = 2;                        // buffer receiving tile kt+2
  for (int kt = 0; kt < NT; ++kt) {
#if defined(USE_TDM)
    if (kt + 2 < NT) {                // async DMA tile kt+2, 2 iters ahead
      const unsigned long long ko = (unsigned long long)(kt + 2) * KB;
      if (wave == 0)      tdm_load_tile(ldsA[ins], gA + ko, (unsigned int)MDIM);
      else if (wave == 1) tdm_load_tile(ldsB[ins], gB + ko, (unsigned int)NDIM);
    }
#else
    if (kt + 1 < NT) {                // prefetch next tile into registers
      const int ko = (kt + 1) * BK;
      ra0 = *(const v8h*)(gA0 + ko);
      ra1 = *(const v8h*)(gA1 + ko);
      rb0 = *(const v8h*)(gB0 + ko);
      rb1 = *(const v8h*)(gB1 + ko);
    }
#endif

    // A fragments: ISA 16-bit A layout — lane<16 holds K {0..7,16..23},
    // lane>=16 holds K {8..15,24..31}, M = lane%16 within the 16-row frag.
    v16h af[4];
#pragma unroll
    for (int f = 0; f < 4; ++f) {
      const _Float16* p = &sA[cur][(wm * 64 + f * 16 + lh) * LDT];
      v8h lo = *(const v8h*)(p + hi * 8);
      v8h hh = *(const v8h*)(p + 16 + hi * 8);
      af[f] = __builtin_shufflevector(lo, hh, 0, 1, 2, 3, 4, 5, 6, 7,
                                      8, 9, 10, 11, 12, 13, 14, 15);
    }
    // B fragments: lane<16 holds K {0..7,16..23}, lane>=16 holds K {8..15,24..31},
    // N = lane%16 within the 16-col frag; B rows are W rows (K-contiguous).
    v16h bf[2];
#pragma unroll
    for (int g = 0; g < 2; ++g) {
      const _Float16* p = &sB[cur][(wn * 32 + g * 16 + lh) * LDT + hi * 8];
      v8h lo = *(const v8h*)p;
      v8h hh = *(const v8h*)(p + 16);
      bf[g] = __builtin_shufflevector(lo, hh, 0, 1, 2, 3, 4, 5, 6, 7,
                                      8, 9, 10, 11, 12, 13, 14, 15);
    }

#pragma unroll
    for (int f = 0; f < 4; ++f)
#pragma unroll
      for (int g = 0; g < 2; ++g)
        acc[f][g] = __builtin_amdgcn_wmma_f32_16x16x32_f16(
            false, af[f], false, bf[g], (short)0, acc[f][g], false, false);

    if (kt + 1 < NT) {
#if defined(USE_TDM)
      // Tile kt+1 must be resident.  While still issuing (kt+2 issued) the
      // newest op may remain in flight -> wait <=1; in the tail drain fully.
      if (kt + 2 < NT) __builtin_amdgcn_s_wait_tensorcnt((short)1);
      else             __builtin_amdgcn_s_wait_tensorcnt((short)0);
      __syncthreads();
#else
      const int nxt = (cur == STG - 1) ? 0 : cur + 1;
      *(v8h*)&sA[nxt][row0 * LDT + kp] = ra0;
      *(v8h*)&sA[nxt][row1 * LDT + kp] = ra1;
      *(v8h*)&sB[nxt][row0 * LDT + kp] = rb0;
      *(v8h*)&sB[nxt][row1 * LDT + kp] = rb1;
      __syncthreads();
#endif
    }
    cur = (cur == STG - 1) ? 0 : cur + 1;
    ins = (ins == STG - 1) ? 0 : ins + 1;
  }

  // C/D layout: VGPR r holds M = r + 8*hi, N = lane%16.
#pragma unroll
  for (int f = 0; f < 4; ++f) {
#pragma unroll
    for (int g = 0; g < 2; ++g) {
      const int rbase = m0 + wm * 64 + f * 16 + hi * 8;
      const int col   = n0 + wn * 32 + g * 16 + lh;
      float* cp = C + (size_t)rbase * NDIM + col;
#pragma unroll
      for (int r = 0; r < 8; ++r)
        cp[(size_t)r * NDIM] = acc[f][g][r];
    }
  }
}

// ---------------------------------------------------------------------------
// Launcher: inputs are (x f32, quant_indices i32, scales f32); out f32.
// Workspace: [0, 32MiB) = W16, [32MiB, 96MiB) = X16.
// ---------------------------------------------------------------------------
extern "C" void kernel_launch(void* const* d_in, const int* in_sizes, int n_in,
                              void* d_out, int out_size, void* d_ws,
                              size_t ws_size, hipStream_t stream) {
  const float* x      = (const float*)d_in[0];
  const int*   qidx   = (const int*)d_in[1];
  const float* scales = (const float*)d_in[2];
  float*       out    = (float*)d_out;

  _Float16* w16 = (_Float16*)d_ws;
  _Float16* x16 = (_Float16*)((char*)d_ws + (size_t)OUT_F * IN_F * sizeof(_Float16));

  // 1) Dequantize NF4 weights to f16 (16384 blocks x 1024 elems).
  nf4_dequant_kernel<<<NQBLK, 256, 0, stream>>>(qidx, scales, w16);

  // 2) Convert activations to f16 (8 elems / thread).
  const int cvt_blocks = (MDIM * KDIM) / (8 * 256);   // 16384
  cvt_f32_to_f16_kernel<<<cvt_blocks, 256, 0, stream>>>(x, x16);

  // 3) WMMA GEMM: grid (N/128, M/128) = (32, 64).
  dim3 grid(NDIM / 128, MDIM / 128);
  nf4_wmma_gemm_kernel<<<grid, 256, 0, stream>>>(x16, w16, out);
}
